// SelfAttentionTP_40931038331122
// MI455X (gfx1250) — matrix operand order repaired and checked by reading, not compiled, not run on hardware
//
#include <hip/hip_runtime.h>
#include <math.h>

// ---------------- types ----------------
typedef _Float16 half_t;
typedef _Float16 v16h __attribute__((ext_vector_type(16)));
typedef _Float16 v8h  __attribute__((ext_vector_type(8)));
typedef float    v8f  __attribute__((ext_vector_type(8)));

#define S_LEN    3072
#define D_DIM    2048
#define N_HEADS  16
#define HEAD_DIM 128
#define ROT_C    64
#define WIN      768     // WINDOW * L_TOKENS = 3 * 256
#define EPS_RMS  1e-6f

__device__ __forceinline__ v8f wmma16x16x32(v16h a, v16h b, v8f c) {
  return __builtin_amdgcn_wmma_f32_16x16x32_f16(
      /*neg_a=*/false, a, /*neg_b=*/false, b,
      /*c_mod=*/(short)0, c, /*reuse_a=*/false, /*reuse_b=*/false);
}

__device__ __forceinline__ v16h join8(v8h lo, v8h hi) {
  union { v16h v; v8h h[2]; } u;
  u.h[0] = lo; u.h[1] = hi;
  return u.v;
}

// =====================================================================
// Kernel 0: bulk f32 -> f16 conversion (for x, weights, V).
// n must be a multiple of 2048; grid = n/2048, 256 threads, 8 elems/thread.
// =====================================================================
__global__ __launch_bounds__(256)
void cvt_f32_to_f16(const float* __restrict__ in, half_t* __restrict__ out)
{
  const size_t i = ((size_t)blockIdx.x * 256 + threadIdx.x) * 8;
  const float4 a = *(const float4*)(in + i);
  const float4 b = *(const float4*)(in + i + 4);
  v8h o;
  o[0] = (half_t)a.x; o[1] = (half_t)a.y; o[2] = (half_t)a.z; o[3] = (half_t)a.w;
  o[4] = (half_t)b.x; o[5] = (half_t)b.y; o[6] = (half_t)b.z; o[7] = (half_t)b.w;
  *(v8h*)(out + i) = o;
}

// =====================================================================
// Kernel 1: C(MxN,f32) = A(MxK,f16) * W(NxK,f16)^T + bias(N).
// 256 threads / 8 waves, 128x128 tile, K step 32, double-buffered LDS,
// software pipelined (global loads of tile k+1 overlap WMMAs of tile k).
// =====================================================================
#define GLDS 40   // padded LDS row stride in halfs (80B -> conflict-free)

__global__ __launch_bounds__(256)
void gemm_h_xwT_bias(const half_t* __restrict__ A, const half_t* __restrict__ W,
                     const float* __restrict__ bias, float* __restrict__ C,
                     int M, int N, int K)
{
  __shared__ __attribute__((aligned(16))) half_t As[2][128 * GLDS];
  __shared__ __attribute__((aligned(16))) half_t Bs[2][128 * GLDS];

  const int tid   = threadIdx.x;
  const int lane  = tid & 31;
  const int wave  = tid >> 5;
  const int l16   = lane & 15;
  const int lhalf = lane >> 4;
  const int wrow  = wave & 3;        // 4 row groups of 32
  const int wcol  = wave >> 2;       // 2 col groups of 64
  const int m0 = blockIdx.y * 128;
  const int n0 = blockIdx.x * 128;
  (void)M;

  // staging chunk assignment: 128 rows x 4 v8h chunks = 512 chunks, 2/thread
  int srow[2], skof[2];
  #pragma unroll
  for (int p = 0; p < 2; ++p) {
    const int id = tid + p * 256;
    srow[p] = id >> 2;
    skof[p] = (id & 3) * 8;
  }

  v8f acc[2][4];
  #pragma unroll
  for (int i = 0; i < 2; ++i)
    #pragma unroll
    for (int j = 0; j < 4; ++j)
      #pragma unroll
      for (int e = 0; e < 8; ++e) acc[i][j][e] = 0.f;

  v8h areg[2], breg[2];
  // prologue: tile 0
  #pragma unroll
  for (int p = 0; p < 2; ++p) {
    areg[p] = *(const v8h*)(A + (size_t)(m0 + srow[p]) * K + skof[p]);
    breg[p] = *(const v8h*)(W + (size_t)(n0 + srow[p]) * K + skof[p]);
  }
  #pragma unroll
  for (int p = 0; p < 2; ++p) {
    *(v8h*)(&As[0][srow[p] * GLDS + skof[p]]) = areg[p];
    *(v8h*)(&Bs[0][srow[p] * GLDS + skof[p]]) = breg[p];
  }
  __syncthreads();

  const int nk = K >> 5;
  for (int ki = 0; ki < nk; ++ki) {
    const int cur = ki & 1;
    const int nxt = cur ^ 1;
    // issue next tile's global loads early
    if (ki + 1 < nk) {
      const int k1 = (ki + 1) << 5;
      #pragma unroll
      for (int p = 0; p < 2; ++p) {
        areg[p] = *(const v8h*)(A + (size_t)(m0 + srow[p]) * K + k1 + skof[p]);
        breg[p] = *(const v8h*)(W + (size_t)(n0 + srow[p]) * K + k1 + skof[p]);
      }
      if (ki + 2 < nk) {
        __builtin_prefetch(A + (size_t)(m0 + srow[0]) * K + ((ki + 2) << 5) + skof[0], 0, 3);
        __builtin_prefetch(W + (size_t)(n0 + srow[0]) * K + ((ki + 2) << 5) + skof[0], 0, 3);
      }
    }

    // fragments from current buffer
    v16h afrag[2];
    #pragma unroll
    for (int rt = 0; rt < 2; ++rt) {
      const half_t* base = &As[cur][(wrow * 32 + rt * 16 + l16) * GLDS];
      afrag[rt] = join8(*(const v8h*)(base + lhalf * 8),
                        *(const v8h*)(base + 16 + lhalf * 8));
    }
    v16h bfrag[4];
    #pragma unroll
    for (int ct = 0; ct < 4; ++ct) {
      const half_t* base = &Bs[cur][(wcol * 64 + ct * 16 + l16) * GLDS + lhalf * 8];
      bfrag[ct] = join8(*(const v8h*)(base), *(const v8h*)(base + 16));
    }
    #pragma unroll
    for (int rt = 0; rt < 2; ++rt)
      #pragma unroll
      for (int ct = 0; ct < 4; ++ct)
        acc[rt][ct] = wmma16x16x32(afrag[rt], bfrag[ct], acc[rt][ct]);

    // stage next tile into the other buffer
    if (ki + 1 < nk) {
      #pragma unroll
      for (int p = 0; p < 2; ++p) {
        *(v8h*)(&As[nxt][srow[p] * GLDS + skof[p]]) = areg[p];
        *(v8h*)(&Bs[nxt][srow[p] * GLDS + skof[p]]) = breg[p];
      }
    }
    __syncthreads();
  }

  // epilogue: C layout lane -> (M=r+8*lhalf, N=l16)
  #pragma unroll
  for (int rt = 0; rt < 2; ++rt) {
    #pragma unroll
    for (int ct = 0; ct < 4; ++ct) {
      const int gn = n0 + wcol * 64 + ct * 16 + l16;
      const float b = bias[gn];
      #pragma unroll
      for (int r = 0; r < 8; ++r) {
        const int gm = m0 + wrow * 32 + rt * 16 + r + lhalf * 8;
        C[(size_t)gm * N + gn] = acc[rt][ct][r] + b;
      }
    }
  }
}

// =====================================================================
// Kernel 2: fused RMSNorm (over D=2048) + RoPE, f32 in -> f16 out.
// =====================================================================
__global__ __launch_bounds__(256)
void rmsnorm_rope(const float* __restrict__ X, const float* __restrict__ g,
                  const float* __restrict__ cosb, const float* __restrict__ sinb,
                  half_t* __restrict__ Y, float outscale)
{
  __shared__ float red[256];
  const int s   = blockIdx.x;
  const int tid = threadIdx.x;
  const float* row = X + (size_t)s * D_DIM;
  const int base = tid * 8;

  float v[8];
  const float4 a = *(const float4*)(row + base);
  const float4 b = *(const float4*)(row + base + 4);
  v[0] = a.x; v[1] = a.y; v[2] = a.z; v[3] = a.w;
  v[4] = b.x; v[5] = b.y; v[6] = b.z; v[7] = b.w;

  float ss = 0.f;
  #pragma unroll
  for (int i = 0; i < 8; ++i) ss += v[i] * v[i];
  red[tid] = ss;
  __syncthreads();
  for (int off = 128; off > 0; off >>= 1) {
    if (tid < off) red[tid] += red[tid + off];
    __syncthreads();
  }
  const float rn = rsqrtf(red[0] / (float)D_DIM + EPS_RMS);

  #pragma unroll
  for (int p = 0; p < 4; ++p) {
    const int d = base + 2 * p;
    const int c = (d & (HEAD_DIM - 1)) >> 1;
    const float xr = v[2 * p]     * rn * g[d];
    const float xi = v[2 * p + 1] * rn * g[d + 1];
    const float cv = cosb[s * ROT_C + c];
    const float sv = sinb[s * ROT_C + c];
    Y[(size_t)s * D_DIM + d]     = (half_t)((xr * cv - xi * sv) * outscale);
    Y[(size_t)s * D_DIM + d + 1] = (half_t)((xr * sv + xi * cv) * outscale);
  }
}

// =====================================================================
// Kernel 3: banded flash attention, f16 in / f16 out.
// WG = 4 waves / 128 threads, one 64-query block per WG (one head).
// Each 32-key K/V tile is staged into LDS ONCE and shared by 4 waves.
// =====================================================================
#define VLDS 40    // VT [col][key] stride in halfs
#define KLDS 136   // Ks [key][col] stride in halfs (272B -> conflict-free b128)

__global__ __launch_bounds__(128)
void attn_window(const half_t* __restrict__ Qh, const half_t* __restrict__ Kh,
                 const half_t* __restrict__ Vh, half_t* __restrict__ O)
{
  __shared__ __attribute__((aligned(16))) half_t Ks[32 * KLDS];       // [key][hd]
  __shared__ __attribute__((aligned(16))) half_t VT[HEAD_DIM * VLDS]; // [hd][key]
  __shared__ __attribute__((aligned(16))) half_t Pt[4][16 * VLDS];    // per wave

  const int tid    = threadIdx.x;
  const int lane   = tid & 31;
  const int wave   = tid >> 5;
  const int l16    = lane & 15;
  const int lhalf  = lane >> 4;
  const int qblock = blockIdx.x * 64;
  const int head   = blockIdx.y;
  const int qbase  = qblock + wave * 16;

  // ---- preload this wave's Q A-fragments (4 k-steps over HEAD_DIM) ----
  v16h qf[4];
  const half_t* qrow = Qh + (size_t)(qbase + l16) * D_DIM + head * HEAD_DIM;
  #pragma unroll
  for (int ks = 0; ks < 4; ++ks) {
    const half_t* p = qrow + ks * 32 + lhalf * 8;
    qf[ks] = join8(*(const v8h*)p, *(const v8h*)(p + 16));
  }

  float mrow[8], lrow[8];
  v8f oacc[8];
  #pragma unroll
  for (int r = 0; r < 8; ++r) { mrow[r] = -1e30f; lrow[r] = 0.f; }
  #pragma unroll
  for (int t = 0; t < 8; ++t)
    #pragma unroll
    for (int e = 0; e < 8; ++e) oacc[t][e] = 0.f;

  int kstart = qblock - WIN; if (kstart < 0) kstart = 0; kstart &= ~31;
  int kend = qblock + 64 + WIN; if (kend > S_LEN) kend = S_LEN;
  // qblock % 64 == 0 and WIN % 32 == 0  =>  kstart/kend are 32-aligned.

  // staging assignment for K: 32 rows x 16 v8h chunks = 512 chunks, 4/thread
  const int vkey  = tid & 31;          // V staging: this thread's key
  const int vcol0 = (tid >> 5) * 32;   // V staging: hd range base

  for (int kb = kstart; kb < kend; kb += 32) {
    // ---- cooperative staging: K tile row-major, V tile transposed ----
    #pragma unroll
    for (int p = 0; p < 4; ++p) {
      const int id  = tid + p * 128;
      const int row = id >> 4;
      const int ko  = (id & 15) * 8;
      const v8h kv = *(const v8h*)(Kh + (size_t)(kb + row) * D_DIM + head * HEAD_DIM + ko);
      *(v8h*)(&Ks[row * KLDS + ko]) = kv;
    }
    {
      const half_t* vrow = Vh + (size_t)(kb + vkey) * D_DIM + head * HEAD_DIM + vcol0;
      #pragma unroll
      for (int j = 0; j < 4; ++j) {
        const v8h vv = *(const v8h*)(vrow + j * 8);
        #pragma unroll
        for (int e = 0; e < 8; ++e)
          VT[(vcol0 + j * 8 + e) * VLDS + vkey] = vv[e];
      }
    }
    __syncthreads();

    // ---- scores: two 16-key tiles, 4 WMMAs each over HEAD_DIM ----
    v8f sacc[2];
    #pragma unroll
    for (int kt = 0; kt < 2; ++kt) {
      #pragma unroll
      for (int e = 0; e < 8; ++e) sacc[kt][e] = 0.f;
      const half_t* kbase = &Ks[(kt * 16 + l16) * KLDS];
      #pragma unroll
      for (int ks = 0; ks < 4; ++ks) {
        const half_t* p = kbase + ks * 32 + lhalf * 8;
        const v16h kf = join8(*(const v8h*)p, *(const v8h*)(p + 16));
        sacc[kt] = wmma16x16x32(qf[ks], kf, sacc[kt]);
      }
    }

    // ---- band mask + online softmax ----
    const int k0g = kb + l16;
    const int k1g = kb + 16 + l16;
    #pragma unroll
    for (int r = 0; r < 8; ++r) {
      const int qg = qbase + r + lhalf * 8;
      float s0 = sacc[0][r];
      float s1 = sacc[1][r];
      int d0 = k0g - qg; if (d0 < 0) d0 = -d0;
      int d1 = k1g - qg; if (d1 < 0) d1 = -d1;
      if (d0 > WIN) s0 = -1e30f;
      if (d1 > WIN) s1 = -1e30f;

      float vmax = fmaxf(s0, s1);
      #pragma unroll
      for (int m = 1; m < 16; m <<= 1) vmax = fmaxf(vmax, __shfl_xor(vmax, m, 32));
      const float mnew = fmaxf(mrow[r], vmax);
      const float corr = expf(mrow[r] - mnew);
      const float p0 = (s0 < -1e29f) ? 0.f : expf(s0 - mnew);
      const float p1 = (s1 < -1e29f) ? 0.f : expf(s1 - mnew);
      float rs = p0 + p1;
      #pragma unroll
      for (int m = 1; m < 16; m <<= 1) rs += __shfl_xor(rs, m, 32);
      lrow[r] = lrow[r] * corr + rs;
      mrow[r] = mnew;
      #pragma unroll
      for (int t = 0; t < 8; ++t) oacc[t][r] *= corr;

      Pt[wave][(r + lhalf * 8) * VLDS + l16]      = (half_t)p0;
      Pt[wave][(r + lhalf * 8) * VLDS + 16 + l16] = (half_t)p1;
    }

    // ---- P (16x32) A-fragment (wave-local LDS, in-order DS), 8 PV WMMAs ----
    const half_t* pbase = &Pt[wave][l16 * VLDS + lhalf * 8];
    const v16h pf = join8(*(const v8h*)pbase, *(const v8h*)(pbase + 16));
    #pragma unroll
    for (int nt = 0; nt < 8; ++nt) {
      const half_t* vb = &VT[(nt * 16 + l16) * VLDS + lhalf * 8];
      const v16h vf = join8(*(const v8h*)vb, *(const v8h*)(vb + 16));
      oacc[nt] = wmma16x16x32(pf, vf, oacc[nt]);
    }
    __syncthreads();   // protect Ks/VT before next staging round
  }

  // ---- normalize and write f16 (feeds O-projection GEMM directly) ----
  #pragma unroll
  for (int nt = 0; nt < 8; ++nt) {
    const int gn = head * HEAD_DIM + nt * 16 + l16;
    #pragma unroll
    for (int r = 0; r < 8; ++r) {
      const int gq = qbase + r + lhalf * 8;
      O[(size_t)gq * D_DIM + gn] = (half_t)(oacc[nt][r] / lrow[r]);
    }
  }
}

// =====================================================================
// Host-side launcher
// =====================================================================
extern "C" void kernel_launch(void* const* d_in, const int* in_sizes, int n_in,
                              void* d_out, int out_size, void* d_ws, size_t ws_size,
                              hipStream_t stream)
{
  (void)in_sizes; (void)n_in; (void)out_size; (void)ws_size;
  const float* x    = (const float*)d_in[0];
  const float* wq   = (const float*)d_in[1];
  const float* bq   = (const float*)d_in[2];
  const float* wk   = (const float*)d_in[3];
  const float* bk   = (const float*)d_in[4];
  const float* wv   = (const float*)d_in[5];
  const float* bv   = (const float*)d_in[6];
  const float* wo   = (const float*)d_in[7];
  const float* bo   = (const float*)d_in[8];
  const float* gq   = (const float*)d_in[9];
  const float* gk   = (const float*)d_in[10];
  const float* cosb = (const float*)d_in[11];
  const float* sinb = (const float*)d_in[12];
  float* out = (float*)d_out;

  char* ws = (char*)d_ws;
  const size_t SD  = (size_t)S_LEN * D_DIM;   // 6291456
  const size_t DD  = (size_t)D_DIM * D_DIM;   // 4194304
  const size_t fS  = SD * sizeof(float);
  const size_t hS  = SD * sizeof(half_t);
  const size_t hW  = DD * sizeof(half_t);

  float*  qraw = (float*)(ws);                 // f32 Q pre-norm   (later: attnh)
  float*  kraw = (float*)(ws + fS);            // f32 K pre-norm   (later: vh)
  float*  vbuf = (float*)(ws + 2 * fS);        // f32 V
  half_t* qh   = (half_t*)(ws + 3 * fS);
  half_t* kh   = (half_t*)(ws + 3 * fS + hS);
  half_t* xh   = (half_t*)(ws + 3 * fS + 2 * hS);
  half_t* wqh  = (half_t*)(ws + 3 * fS + 3 * hS);
  half_t* wkh  = (half_t*)(ws + 3 * fS + 3 * hS + hW);
  half_t* wvh  = (half_t*)(ws + 3 * fS + 3 * hS + 2 * hW);
  half_t* woh  = (half_t*)(ws + 3 * fS + 3 * hS + 3 * hW);
  half_t* attnh = (half_t*)qraw;               // alias: qraw dead after rope(Q)
  half_t* vh    = (half_t*)kraw;               // alias: kraw dead after rope(K)

  // ---- pre-convert activations + weights to f16 ----
  cvt_f32_to_f16<<<(int)(SD / 2048), 256, 0, stream>>>(x,  xh);
  cvt_f32_to_f16<<<(int)(DD / 2048), 256, 0, stream>>>(wq, wqh);
  cvt_f32_to_f16<<<(int)(DD / 2048), 256, 0, stream>>>(wk, wkh);
  cvt_f32_to_f16<<<(int)(DD / 2048), 256, 0, stream>>>(wv, wvh);
  cvt_f32_to_f16<<<(int)(DD / 2048), 256, 0, stream>>>(wo, woh);

  // ---- QKV projections ----
  const dim3 ggrid(D_DIM / 128, S_LEN / 128);
  gemm_h_xwT_bias<<<ggrid, 256, 0, stream>>>(xh, wqh, bq, qraw, S_LEN, D_DIM, D_DIM);
  gemm_h_xwT_bias<<<ggrid, 256, 0, stream>>>(xh, wkh, bk, kraw, S_LEN, D_DIM, D_DIM);
  gemm_h_xwT_bias<<<ggrid, 256, 0, stream>>>(xh, wvh, bv, vbuf, S_LEN, D_DIM, D_DIM);

  // ---- RMSNorm + RoPE (softmax scale folded into Q), V -> f16 ----
  const float qscale = 1.0f / sqrtf((float)HEAD_DIM);
  rmsnorm_rope<<<S_LEN, 256, 0, stream>>>(qraw, gq, cosb, sinb, qh, qscale);
  rmsnorm_rope<<<S_LEN, 256, 0, stream>>>(kraw, gk, cosb, sinb, kh, 1.0f);
  cvt_f32_to_f16<<<(int)(SD / 2048), 256, 0, stream>>>(vbuf, vh);

  // ---- banded flash attention ----
  attn_window<<<dim3(S_LEN / 64, N_HEADS), 128, 0, stream>>>(qh, kh, vh, attnh);

  // ---- output projection ----
  gemm_h_xwT_bias<<<ggrid, 256, 0, stream>>>(attnh, woh, bo, out, S_LEN, D_DIM, D_DIM);
}
